// SSRupsampling3_44470091383395
// MI455X (gfx1250) — matrix operand order repaired and checked by reading, not compiled, not run on hardware
//
#include <hip/hip_runtime.h>
#include <hip/hip_bf16.h>

// ---------------------------------------------------------------------------
// SSRupsampling3 on MI455X (gfx1250).
//   GEMM1: Y[32768,384] = X[32768,512] * Wcat^T   (+bias/relu on cols>=128)
//   GEMM2: Z[32768,512] = Y * Wbig^T  -> pixel-shuffle scatter to out
// Operands live in WMMA-fragment-packed layout (32B/lane/fragment): fragment
// loads are 2x ds_load_b128; A/B staging is contiguous 16KB copies via
// GLOBAL_LOAD_ASYNC_TO_LDS_B128; X conversion uses v_cvt_pk_bf16_f32.
// Workspace: WcatS(384KB) WbigS(384KB) Wf(384KB) Yswz(24MB)  ~26.4MB
// ---------------------------------------------------------------------------

typedef __attribute__((ext_vector_type(16))) __bf16 v16bf;
typedef __attribute__((ext_vector_type(2)))  __bf16 v2bf;
typedef __attribute__((ext_vector_type(8)))  float  v8f;
typedef int v4i __attribute__((vector_size(16)));   // matches async builtin pointee

#define CIN   512
#define NPOS  32768
#define NCOL1 384
#define NCOL2 512
#define KDIM2 384

#if defined(__gfx1250__) && __has_builtin(__builtin_amdgcn_global_load_async_to_lds_b128) && __has_builtin(__builtin_amdgcn_s_wait_asynccnt)
#define HAVE_ASYNC_LDS 1
#else
#define HAVE_ASYNC_LDS 0
#endif

#if defined(__gfx1250__) && __has_builtin(__builtin_amdgcn_cvt_pk_bf16_f32)
#define HAVE_PK_BF16 1
#else
#define HAVE_PK_BF16 0
#endif

__device__ __forceinline__ unsigned short f2bf(float f) {
  unsigned u = __float_as_uint(f);
  u += 0x7FFFu + ((u >> 16) & 1u);       // round-to-nearest-even
  return (unsigned short)(u >> 16);
}

// pack two f32 -> dword of two bf16 (lo, hi); hardware v_cvt_pk_bf16_f32 if present
__device__ __forceinline__ unsigned pkbf(float lo, float hi) {
#if HAVE_PK_BF16
  v2bf p = __builtin_amdgcn_cvt_pk_bf16_f32(lo, hi);
  return __builtin_bit_cast(unsigned, p);
#else
  return (unsigned)f2bf(lo) | ((unsigned)f2bf(hi) << 16);
#endif
}

union FragBF { v16bf v; uint4 q[2]; };

// Fragment-packed addressing (ushort index within a 16KB chunk).
// B chunk = 64K x 128N: frag f = (k>>5)*8 + (n>>4); lane = ((k>>3)&1)*16+(n&15)
__device__ __forceinline__ int bswz(int kin, int n) {
  const int f  = ((kin >> 5) << 3) + (n >> 4);
  const int ln = (((kin >> 3) & 1) << 4) + (n & 15);
  const int v  = ((kin & 7) >> 1) + (((kin >> 4) & 1) << 2);
  return f * 512 + ln * 16 + v * 2 + (kin & 1);
}
// A chunk = 128M x 64K, ISA 16x32 A layout (half/v interleave)
__device__ __forceinline__ int aswz(int m, int kin) {
  const int kt = kin >> 5, kk = kin & 31, p = kk >> 1;
  const int half = (p >> 2) & 1;
  const int v    = (p & 3) + ((p >> 3) << 2);
  const int af   = ((m >> 4) << 1) + kt;
  const int ln   = (half << 4) + (m & 15);
  return af * 512 + ln * 16 + v * 2 + (kin & 1);
}

// Each lane owns 32 contiguous bytes of its fragment: two b128 loads, no repack.
__device__ __forceinline__ v16bf load_frag(const unsigned short* fragBase, int lane) {
  const uint4* p = (const uint4*)fragBase + (lane << 1);
  FragBF f;
  f.q[0] = p[0];
  f.q[1] = p[1];
  return f.v;
}

// Contiguous 16KB global->LDS copy (256 threads). Async path on gfx1250.
__device__ __forceinline__ void copy16k(unsigned short* dstLds,
                                        const unsigned short* srcGlobal, int tid) {
#if HAVE_ASYNC_LDS
#pragma unroll
  for (int it = 0; it < 4; ++it) {
    const int idx = tid + it * 256;      // 16B units
    __builtin_amdgcn_global_load_async_to_lds_b128(
        (__attribute__((address_space(1))) v4i*)(srcGlobal + idx * 8),
        (__attribute__((address_space(3))) v4i*)(dstLds + idx * 8), 0, 0);
  }
#else
  const uint4* s = (const uint4*)srcGlobal;
  uint4* d = (uint4*)dstLds;
#pragma unroll
  for (int it = 0; it < 4; ++it) d[tid + it * 256] = s[tid + it * 256];
#endif
}

__device__ __forceinline__ void async_join() {
#if HAVE_ASYNC_LDS
  __builtin_amdgcn_s_wait_asynccnt(0);
#endif
}

// ---------------------- weight prep (tiny, f32) ----------------------------
__global__ void prep_wcat(const float* __restrict__ w1, const float* __restrict__ wlr,
                          const float* __restrict__ wdu, unsigned short* __restrict__ WcatS) {
  int idx = blockIdx.x * blockDim.x + threadIdx.x;
  if (idx >= CIN * NCOL1) return;
  int i = idx / NCOL1, o = idx % NCOL1;
  float v;
  if (o < 128)      v = w1[o * CIN + i];                         // 1x1
  else if (o < 256) v = wlr[(o - 128) * (CIN * 3) + i * 3 + 1];  // center tap (1,3)
  else              v = wdu[(o - 256) * (CIN * 3) + i * 3 + 1];  // center tap (3,1)
  const int chunk = i >> 6, kin = i & 63, nT = o >> 7, n = o & 127;
  WcatS[(chunk * 3 + nT) * 8192 + bswz(kin, n)] = f2bf(v);
}

__global__ void prep_wfused(const float* __restrict__ w_w, const float* __restrict__ w_ch,
                            float* __restrict__ Wf) {            // Wf[256][384] = w_w @ w_ch
  int idx = blockIdx.x * blockDim.x + threadIdx.x;
  if (idx >= 256 * NCOL1) return;
  int m = idx / NCOL1, t = idx % NCOL1;
  float s = 0.f;
  for (int j = 0; j < 128; ++j) s += w_w[m * 128 + j] * w_ch[j * NCOL1 + t];
  Wf[idx] = s;
}

__global__ void prep_wbig(const float* __restrict__ w_H, const float* __restrict__ Wf,
                          unsigned short* __restrict__ WbigS) {  // Wbig[512][384] -> swizzled
  int idx = blockIdx.x * blockDim.x + threadIdx.x;
  if (idx >= KDIM2 * NCOL2) return;
  int t = idx / NCOL2, u = idx % NCOL2;
  int sblk = u >> 8, r = u & 255;
  float s = 0.f;
  for (int k = 0; k < 128; ++k)
    s += w_H[r * 128 + k] * Wf[(sblk * 128 + k) * NCOL1 + t];
  const int chunk = t >> 6, kin = t & 63, nT = u >> 7, n = u & 127;
  WbigS[(chunk * 4 + nT) * 8192 + bswz(kin, n)] = f2bf(s);
}

// ---------------------- GEMM1: X(f32) x WcatS -> Yswz (bf16, frag-packed) --
__global__ __launch_bounds__(256) void gemm1_kernel(
    const float* __restrict__ X, const unsigned short* __restrict__ WcatS,
    const float* __restrict__ b_lr, const float* __restrict__ b_du,
    unsigned short* __restrict__ Yswz) {
  __shared__ __align__(16) unsigned short sbuf[16384];   // 32KB: A | B
  unsigned short* As = sbuf;
  unsigned short* Bs = sbuf + 8192;
  const int tid = threadIdx.x, lane = tid & 31, wave = tid >> 5;
  const int waveM = wave & 3, waveN = wave >> 2;
  const int pT = blockIdx.x, nT = blockIdx.y;
  const int p0 = pT * 128;                               // 128 | 4096: never crosses image
  const float* Xb = X + (size_t)(p0 >> 12) * (CIN * 4096) + (p0 & 4095);
  // A-staging decomposition: thread owns 4 consecutive pixels x 8 consecutive k
  const int mq = tid & 31;                               // pixel quad: m = mq*4..mq*4+3
  const int g  = tid >> 5;                               // k-group 0..7: kbase = g*8
  const int kt = g >> 2, hi = (g >> 1) & 1, half = g & 1;

  v8f acc[2][4] = {};

  for (int kc = 0; kc < CIN / 64; ++kc) {
    __syncthreads();
    copy16k(Bs, WcatS + (size_t)(kc * 3 + nT) * 8192, tid);
    // A: 8x float4 (vectorized along pixels) -> register transpose ->
    //    v_cvt_pk_bf16_f32 -> 4x ds_store_b128 in fragment layout
    {
      const float* src = Xb + (size_t)(kc * 64 + g * 8) * 4096 + mq * 4;
      float4 cvec[8];
#pragma unroll
      for (int e = 0; e < 8; ++e)
        cvec[e] = *(const float4*)(src + (size_t)e * 4096);
      const float* cf = (const float*)cvec;              // cf[e*4 + mm]
#pragma unroll
      for (int mm = 0; mm < 4; ++mm) {
        const int m = mq * 4 + mm;
        const unsigned d0 = pkbf(cf[0 * 4 + mm], cf[1 * 4 + mm]);
        const unsigned d1 = pkbf(cf[2 * 4 + mm], cf[3 * 4 + mm]);
        const unsigned d2 = pkbf(cf[4 * 4 + mm], cf[5 * 4 + mm]);
        const unsigned d3 = pkbf(cf[6 * 4 + mm], cf[7 * 4 + mm]);
        const int af = ((m >> 4) << 1) + kt;
        const int ln = (half << 4) + (m & 15);
        ((uint4*)As)[af * 64 + ln * 2 + hi] = make_uint4(d0, d1, d2, d3);
      }
    }
    if (kc + 1 < CIN / 64)
      __builtin_prefetch(Xb + (size_t)((kc + 1) * 64) * 4096 + tid * 16, 0, 0);
    async_join();
    __syncthreads();
#pragma unroll
    for (int ks = 0; ks < 2; ++ks) {
      FragBF a[2], b[4];
#pragma unroll
      for (int i = 0; i < 2; ++i)
        a[i].v = load_frag(As + ((((waveM * 2 + i) << 1) + ks) << 9), lane);
#pragma unroll
      for (int j = 0; j < 4; ++j)
        b[j].v = load_frag(Bs + (((ks << 3) + waveN * 4 + j) << 9), lane);
#pragma unroll
      for (int i = 0; i < 2; ++i)
#pragma unroll
        for (int j = 0; j < 4; ++j)
          acc[i][j] = __builtin_amdgcn_wmma_f32_16x16x32_bf16(
              false, a[i].v, false, b[j].v, (short)0, acc[i][j], false, false);
    }
  }

  // epilogue: bias+relu -> frag-swizzled LDS -> contiguous 32KB store to Yswz
  __syncthreads();
#pragma unroll
  for (int i = 0; i < 2; ++i) {
#pragma unroll
    for (int j = 0; j < 4; ++j) {
      const int kl = waveN * 64 + j * 16 + (lane & 15);  // col in 128-tile
      const int o  = nT * 128 + kl;
      const bool act = (o >= 128);
      const float bias = (o >= 256) ? b_du[o - 256] : (act ? b_lr[o - 128] : 0.f);
      const int ch = kl >> 6, kk = kl & 63;
#pragma unroll
      for (int r = 0; r < 8; ++r) {
        float v = acc[i][j][r];
        if (act) v = fmaxf(v + bias, 0.f);
        const int m_loc = waveM * 32 + i * 16 + ((lane >> 4) << 3) + r;
        sbuf[ch * 8192 + aswz(m_loc, kk)] = f2bf(v);
      }
    }
  }
  __syncthreads();
  {
    const uint4* s = (const uint4*)sbuf;
    uint4* d = (uint4*)(Yswz + (size_t)(pT * 6 + nT * 2) * 8192);
#pragma unroll
    for (int it = 0; it < 8; ++it) d[tid + it * 256] = s[tid + it * 256];
  }
}

// ---------------------- GEMM2: Yswz x WbigS -> out (pixel-shuffle scatter) -
__global__ __launch_bounds__(256) void gemm2_kernel(
    const unsigned short* __restrict__ Yswz, const unsigned short* __restrict__ WbigS,
    float* __restrict__ out) {
  __shared__ __align__(16) unsigned short sbuf[16384];
  unsigned short* As = sbuf;
  unsigned short* Bs = sbuf + 8192;
  const int tid = threadIdx.x, lane = tid & 31, wave = tid >> 5;
  const int waveM = wave & 3, waveN = wave >> 2;
  const int pT = blockIdx.x, uT = blockIdx.y;            // uT: 0..3

  v8f acc[2][4] = {};

  for (int kc = 0; kc < KDIM2 / 64; ++kc) {
    __syncthreads();
    copy16k(As, Yswz  + (size_t)(pT * 6 + kc) * 8192, tid);
    copy16k(Bs, WbigS + (size_t)(kc * 4 + uT) * 8192, tid);
    if (kc + 1 < KDIM2 / 64)
      __builtin_prefetch(Yswz + (size_t)(pT * 6 + kc + 1) * 8192 + tid * 32, 0, 0);
    async_join();
    __syncthreads();
#pragma unroll
    for (int ks = 0; ks < 2; ++ks) {
      FragBF a[2], b[4];
#pragma unroll
      for (int i = 0; i < 2; ++i)
        a[i].v = load_frag(As + ((((waveM * 2 + i) << 1) + ks) << 9), lane);
#pragma unroll
      for (int j = 0; j < 4; ++j)
        b[j].v = load_frag(Bs + (((ks << 3) + waveN * 4 + j) << 9), lane);
#pragma unroll
      for (int i = 0; i < 2; ++i)
#pragma unroll
        for (int j = 0; j < 4; ++j)
          acc[i][j] = __builtin_amdgcn_wmma_f32_16x16x32_bf16(
              false, a[i].v, false, b[j].v, (short)0, acc[i][j], false, false);
    }
  }

  // u = uT*128 + c ; out[n, c, 2h+s, 2w+s2] = Z[p,u]
  const int s  = uT >> 1;
  const int s2 = uT & 1;
  const int mrow0 = pT * 128 + waveM * 32 + ((lane >> 4) << 3);
  const int c0    = waveN * 64 + (lane & 15);
#pragma unroll
  for (int i = 0; i < 2; ++i) {
#pragma unroll
    for (int j = 0; j < 4; ++j) {
      const int c = c0 + j * 16;
#pragma unroll
      for (int r = 0; r < 8; ++r) {
        const int p = mrow0 + i * 16 + r;
        const int nImg = p >> 12, hw = p & 4095, h = hw >> 6, w = hw & 63;
        const size_t oidx =
            (((size_t)(nImg * 128 + c) * 128) + (2 * h + s)) * 128 + (2 * w + s2);
        out[oidx] = acc[i][j][r];
      }
    }
  }
}

// ---------------------------------------------------------------------------
extern "C" void kernel_launch(void* const* d_in, const int* in_sizes, int n_in,
                              void* d_out, int out_size, void* d_ws, size_t ws_size,
                              hipStream_t stream) {
  const float* x     = (const float*)d_in[0];
  const float* w_du  = (const float*)d_in[1];
  const float* b_du  = (const float*)d_in[2];
  const float* w_lr  = (const float*)d_in[3];
  const float* b_lr  = (const float*)d_in[4];
  const float* w_1x1 = (const float*)d_in[5];
  const float* w_ch  = (const float*)d_in[6];
  const float* w_w   = (const float*)d_in[7];
  const float* w_H   = (const float*)d_in[8];
  float* out = (float*)d_out;

  char* ws = (char*)d_ws;
  unsigned short* WcatS = (unsigned short*)(ws);             // 8*3  *16KB = 384KB
  unsigned short* WbigS = (unsigned short*)(ws + 393216);    // 6*4  *16KB = 384KB
  float*          Wf    = (float*)(ws + 786432);             // 256*384*4  = 384KB
  unsigned short* Yswz  = (unsigned short*)(ws + 1179648);   // 256*6*16KB = 24MB

  prep_wcat  <<<(CIN * NCOL1 + 255) / 256, 256, 0, stream>>>(w_1x1, w_lr, w_du, WcatS);
  prep_wfused<<<(256 * NCOL1 + 255) / 256, 256, 0, stream>>>(w_w, w_ch, Wf);
  prep_wbig  <<<(KDIM2 * NCOL2 + 255) / 256, 256, 0, stream>>>(w_H, Wf, WbigS);
  gemm1_kernel<<<dim3(NPOS / 128, NCOL1 / 128), 256, 0, stream>>>(x, WcatS, b_lr, b_du, Yswz);
  gemm2_kernel<<<dim3(NPOS / 128, NCOL2 / 128), 256, 0, stream>>>(Yswz, WbigS, out);
}
